// Reslearner_38981123178813
// MI455X (gfx1250) — matrix operand order repaired and checked by reading, not compiled, not run on hardware
//
#include <hip/hip_runtime.h>
#include <hip/hip_bf16.h>
#include <math.h>

// ---------------- CDNA5 WMMA plumbing ----------------
typedef _Float16 h16;
typedef __attribute__((ext_vector_type(16))) _Float16 v16h;
typedef __attribute__((ext_vector_type(8)))  float    v8f;
typedef __attribute__((ext_vector_type(4)))  unsigned int v4u;
typedef __attribute__((ext_vector_type(8)))  unsigned int v8u;

__device__ __forceinline__ v8f wmma16(v16h a, v16h b, v8f c) {
    // D = A(16x32 f16) x B(32x16 f16) + C(16x16 f32)
    return __builtin_amdgcn_wmma_f32_16x16x32_f16(
        /*neg_a=*/false, a, /*neg_b=*/false, b,
        /*c_mod=*/(short)0, c, /*reuse_a=*/false, /*reuse_b=*/false);
}

// A-fragment k pattern (16-bit A 16x32, ISA 7.12.2): lane<16 half=0, lane>=16 half=1
__device__ __forceinline__ int kA(int v, int half) {
    return ((v & 3) << 1) + (half << 3) + ((v >> 2) << 4);
}
// B-fragment k pattern (16-bit B 32x16): same as A, VGPR v holds K=2(v&3)+8*half+16(v>>2), +1
__device__ __forceinline__ int kB(int v, int half) {
    return ((v & 3) << 1) + (half << 3) + ((v >> 2) << 4);
}

// ---------------- problem constants ----------------
#define BSZ    2
#define FEATC  128
#define HLR    64
#define WLR    64
#define KK     4
#define NF     32
#define LIB    345
#define LIBP   352      // padded to 32
#define SPE    224
#define HALF_OFF 14841

// per-layer param offsets within one half of conv3 output
__device__ __constant__ int d_NCI[5] = {8, 32, 32, 32, 32};
// WSTART / BSTART computed from CHANNELS=[8,32,32,32,32,345]
__device__ __constant__ int d_WST[5] = {32, 320, 1376, 2432, 3801};
__device__ __constant__ int d_BST[5] = {0, 288, 1344, 2400, 3456};

// workspace layout (bytes)
#define WS_XFEAT_BYTES ((size_t)BSZ * NF * HLR * WLR * 4)           // 1 MiB
#define WS_IMG_BYTES   ((size_t)BSZ * SPE * 256 * 256 * 4)          // 112 MiB
#define WS_BSPE_HALVES ((size_t)14 * 11 * 32 * 16)                  // 78848
#define WS_BABUN_HALVES ((size_t)16 * 8 * 32 * 16)                  // 65536

// ---------------- hypernetwork param generation ----------------
__device__ __forceinline__ float genparam(const float* __restrict__ conv3_w,
                                          const float* __restrict__ conv3_b,
                                          const float* __restrict__ xf, int row) {
    float acc = conv3_b[row];
    const float* wr = conv3_w + (size_t)row * 32;
#pragma unroll
    for (int c = 0; c < 32; ++c) acc = fmaf(wr[c], xf[c], acc);
    return acc > 0.f ? acc : 0.f;   // relu (lr_params are relu'd)
}

// constant positional-encoding tile matrix T[p=dy*4+dx][c], values in {-1,0,1}
__device__ __forceinline__ float Tval(int p, int c) {
    int dy = p >> 2, dx = p & 3;
    const float cq[4] = {1.f, 0.f, -1.f, 0.f};
    const float sq[4] = {0.f, 1.f, 0.f, -1.f};
    switch (c) {
        case 0: return cq[dx];
        case 1: return sq[dx];
        case 2: return cq[dy];
        case 3: return sq[dy];
        case 4: return (dx & 1) ? -1.f : 1.f;
        case 5: return 0.f;
        case 6: return (dy & 1) ? -1.f : 1.f;
        default: return 0.f;
    }
}

// A-frag from row-major f16 LDS matrix (row m, leading dim ld)
__device__ __forceinline__ v16h loadA_lds(const h16* M, int m, int half, int ld) {
    v16h a;
#pragma unroll
    for (int v = 0; v < 8; ++v) {
        int k = kA(v, half);
        a[2 * v]     = M[m * ld + k];
        a[2 * v + 1] = M[m * ld + k + 1];
    }
    return a;
}
// B-frag from "transposed" f16 LDS weights Wt[n][k], ld = 32
__device__ __forceinline__ v16h loadB_ldsT(const h16* Wt, int n, int half) {
    v16h b;
#pragma unroll
    for (int v = 0; v < 8; ++v) {
        int k = kB(v, half);
        b[2 * v]     = Wt[n * 32 + k];
        b[2 * v + 1] = Wt[n * 32 + k + 1];
    }
    return b;
}
// A-frag from row-major f32 global matrix, converting to f16
__device__ __forceinline__ v16h loadA_f32(const float* __restrict__ M, int m, int half,
                                          int ld, int kbase) {
    v16h a;
#pragma unroll
    for (int v = 0; v < 8; ++v) {
        int k = kbase + kA(v, half);
        a[2 * v]     = (_Float16)M[(size_t)m * ld + k];
        a[2 * v + 1] = (_Float16)M[(size_t)m * ld + k + 1];
    }
    return a;
}

// ---------------- kernel 1: conv1 3x3 reflect-pad + relu ----------------
__global__ __launch_bounds__(256) void conv1_kernel(const float* __restrict__ in,
                                                    const float* __restrict__ cw,
                                                    const float* __restrict__ cb,
                                                    float* __restrict__ xfeat) {
    int idx = blockIdx.x * 256 + threadIdx.x;      // [b][o][h][w], 262144 total
    int wq = idx & 63; int t = idx >> 6;
    int hq = t & 63;   t >>= 6;
    int o  = t & 31;   int b = t >> 5;
    float acc = cb[o];
    for (int c = 0; c < FEATC; ++c) {
        const float* ip = in + (((size_t)b * FEATC + c) * HLR) * WLR;
        const float* wp = cw + ((size_t)o * FEATC + c) * 9;
#pragma unroll
        for (int ky = 0; ky < 3; ++ky) {
            int y = hq + ky - 1;
            y = (y < 0) ? 1 : (y > 63 ? 62 : y);   // reflect pad 1
#pragma unroll
            for (int kx = 0; kx < 3; ++kx) {
                int x = wq + kx - 1;
                x = (x < 0) ? 1 : (x > 63 ? 62 : x);
                acc = fmaf(ip[y * WLR + x], wp[ky * 3 + kx], acc);
            }
        }
    }
    xfeat[idx] = acc > 0.f ? acc : 0.f;
}

// ---------------- kernel 2a: swizzle spe_lib^T into WMMA B-fragments ----------------
// B[l][s] = spe_lib[s][l], K padded 345->352, layout [nt(14)][kt(11)][lane(32)][16 halves]
__global__ __launch_bounds__(256) void spe_swizzle_kernel(const float* __restrict__ spe_lib,
                                                          h16* __restrict__ Bspe) {
    int i = blockIdx.x * 256 + threadIdx.x;
    if (i >= (int)WS_BSPE_HALVES) return;
    int e = i & 15; int t = i >> 4;
    int lane = t & 31; t >>= 5;
    int kt = t % 11; int nt = t / 11;
    int v = e >> 1, j = e & 1, half = lane >> 4;
    int n = (lane & 15) + nt * 16;                 // s index, < 224
    int k = kt * 32 + kB(v, half) + j;             // l index
    float val = (k < LIB) ? spe_lib[(size_t)n * LIB + k] : 0.f;
    Bspe[i] = (_Float16)val;
}

// ---------------- kernel 2b: swizzle abun into WMMA B-fragments ----------------
// B[w][x] = abun[w][x], layout [nt(16)][kt(8)][lane(32)][16 halves]
__global__ __launch_bounds__(256) void abun_swizzle_kernel(const float* __restrict__ abun,
                                                           h16* __restrict__ Babun) {
    int i = blockIdx.x * 256 + threadIdx.x;        // 65536 exactly
    int e = i & 15; int t = i >> 4;
    int lane = t & 31; t >>= 5;
    int kt = t & 7; int nt = t >> 3;
    int v = e >> 1, j = e & 1, half = lane >> 4;
    int n = (lane & 15) + nt * 16;
    int k = kt * 32 + kB(v, half) + j;
    Babun[i] = (_Float16)abun[(size_t)k * 256 + n];
}

// ---------------- kernel 3: per-tile hypernet MLP pass (WMMA everywhere) ----------------
__global__ __launch_bounds__(128) void mlp_pass_kernel(const float* __restrict__ xfeat,
                                                       const float* __restrict__ conv3_w,
                                                       const float* __restrict__ conv3_b,
                                                       const h16* __restrict__ Bspe,
                                                       float* __restrict__ img,
                                                       int offset) {
    __shared__ float xf[32];
    __shared__ float bvec[LIBP];
    __shared__ h16   Wt[LIBP * 32];        // transposed weights Wt[n][k], f16
    __shared__ h16   Hbuf[2][16 * 32];     // ping-pong activations, f16
    __shared__ h16   Az[16 * LIBP];        // last-layer output (after 0.1*tanh), f16

    int tile = blockIdx.x;                 // 0..8191
    int b = tile >> 12;
    int hw = tile & 4095; int h = hw >> 6, w = hw & 63;
    int tid = threadIdx.x;
    int lane = tid & 31, wave = tid >> 5;
    int half = lane >> 4, nlo = lane & 15;

    // warm L2 for the dominant layer-4 hypernet weight region (global_prefetch_b8)
    __builtin_prefetch(conv3_w + ((size_t)(d_WST[4] + offset)) * 32 + (size_t)tid * 64, 0, 0);

    if (tid < 32) xf[tid] = xfeat[(((size_t)b * NF + tid) * HLR + h) * WLR + w];
    __syncthreads();

    // ---- hidden layers 0..3 (nco = 32) ----
    for (int l = 0; l < 4; ++l) {
        int wst = d_WST[l] + offset, bst = d_BST[l] + offset;
        int nci = d_NCI[l];
        for (int idx = tid; idx < 32 * 32; idx += 128) {
            int n = idx >> 5, k = idx & 31;
            float v = (k < nci) ? genparam(conv3_w, conv3_b, xf, wst + k * 32 + n) : 0.f;
            Wt[n * 32 + k] = (_Float16)v;
        }
        for (int n = tid; n < 32; n += 128)
            bvec[n] = genparam(conv3_w, conv3_b, xf, bst + n);
        __syncthreads();

        if (wave < 2) {
            int nt = wave, n = nlo + 16 * nt;
            v16h a;
            if (l == 0) {
                // constant positional-encoding tile, K padded 8->32 with zeros
#pragma unroll
                for (int v = 0; v < 8; ++v) {
#pragma unroll
                    for (int j = 0; j < 2; ++j) {
                        int k = kA(v, half) + j;
                        a[2 * v + j] = (_Float16)((k < 8) ? Tval(nlo, k) : 0.f);
                    }
                }
            } else {
                a = loadA_lds(Hbuf[(l + 1) & 1], nlo, half, 32);
            }
            v16h bf = loadB_ldsT(Wt, n, half);
            v8f acc = {};
            acc = wmma16(a, bf, acc);
            h16* Hn = Hbuf[l & 1];
#pragma unroll
            for (int v = 0; v < 8; ++v) {
                float x = acc[v] + bvec[n];
                x = x >= 0.f ? x : 0.01f * x;          // leaky relu
                Hn[(v + 8 * half) * 32 + n] = (_Float16)x;
            }
        }
        __syncthreads();
    }

    // ---- layer 4: 32 -> 345 (padded 352), activation 0.1*tanh ----
    {
        int wst = d_WST[4] + offset, bst = d_BST[4] + offset;
        for (int idx = tid; idx < LIBP * 32; idx += 128) {
            int n = idx >> 5, k = idx & 31;
            float v = (n < LIB) ? genparam(conv3_w, conv3_b, xf, wst + k * LIB + n) : 0.f;
            Wt[n * 32 + k] = (_Float16)v;
        }
        for (int n = tid; n < LIBP; n += 128)
            bvec[n] = (n < LIB) ? genparam(conv3_w, conv3_b, xf, bst + n) : 0.f;
        __syncthreads();

        v16h a = loadA_lds(Hbuf[1], nlo, half, 32);    // layer 3 wrote Hbuf[1]
        for (int nt = wave; nt < 22; nt += 4) {
            int n = nlo + 16 * nt;
            v16h bf = loadB_ldsT(Wt, n, half);
            v8f acc = {};
            acc = wmma16(a, bf, acc);
#pragma unroll
            for (int v = 0; v < 8; ++v) {
                float z = 0.1f * tanhf(acc[v] + bvec[n]);
                Az[(v + 8 * half) * LIBP + n] = (_Float16)z;
            }
        }
        __syncthreads();
    }

    // ---- spectral projection: [16x352] x [352x224] via pre-swizzled Bspe ----
    for (int nt = wave; nt < 14; nt += 4) {
        v8f acc = {};
        for (int kt = 0; kt < 11; ++kt) {
            v16h a;
#pragma unroll
            for (int v = 0; v < 8; ++v) {
                int k = kt * 32 + kA(v, half);
                a[2 * v]     = Az[nlo * LIBP + k];
                a[2 * v + 1] = Az[nlo * LIBP + k + 1];
            }
            v16h bf = *(const v16h*)(Bspe + (((size_t)nt * 11 + kt) * 32 + lane) * 16);
            acc = wmma16(a, bf, acc);
        }
        int s = nt * 16 + nlo;
#pragma unroll
        for (int v = 0; v < 8; ++v) {
            int p = v + 8 * half;
            int dy = p >> 2, dx = p & 3;
            img[(((size_t)b * SPE + s) * 256 + (h * 4 + dy)) * 256 + (w * 4 + dx)] = acc[v];
        }
    }
}

// ---------------- kernel 4: ref2 = img1 @ abun (per (b,c) 256x256x256 GEMM) ----------------
__global__ __launch_bounds__(256) void ref2_kernel(const float* __restrict__ img1,
                                                   const h16* __restrict__ Babun,
                                                   float* __restrict__ outb) {
    int gw = (blockIdx.x * 256 + threadIdx.x) >> 5;   // global wave: 448*16*4
    int lane = threadIdx.x & 31;
    int half = lane >> 4, nlo = lane & 15;
    int nq = gw & 3; int t = gw >> 2;
    int mt = t & 15; int slice = t >> 4;              // 0..447
    const float* A = img1 + (size_t)slice * 65536;
    float* O = outb + (size_t)slice * 65536;
    int m = mt * 16 + nlo;

    v8f acc0 = {}, acc1 = {}, acc2 = {}, acc3 = {};
    for (int kt = 0; kt < 8; ++kt) {
        v16h a = loadA_f32(A, m, half, 256, kt * 32);
        const h16* bb = Babun + (((size_t)(nq * 4) * 8 + kt) * 32 + lane) * 16;
        const size_t ntStride = (size_t)8 * 32 * 16;  // halves between nt fragments
        v16h b0 = *(const v16h*)(bb);
        v16h b1 = *(const v16h*)(bb + ntStride);
        v16h b2 = *(const v16h*)(bb + 2 * ntStride);
        v16h b3 = *(const v16h*)(bb + 3 * ntStride);
        acc0 = wmma16(a, b0, acc0);
        acc1 = wmma16(a, b1, acc1);
        acc2 = wmma16(a, b2, acc2);
        acc3 = wmma16(a, b3, acc3);
    }
#pragma unroll
    for (int v = 0; v < 8; ++v) {
        int row = mt * 16 + v + 8 * half;
        size_t base = (size_t)row * 256 + (size_t)(nq * 4) * 16 + nlo;
        O[base]      = acc0[v];
        O[base + 16] = acc1[v];
        O[base + 32] = acc2[v];
        O[base + 48] = acc3[v];
    }
}

// ---------------- kernel 5: out = ref2 + img2 @ ref2 (per (b,c) slice) ----------------
// One WG per (slice, column-half). The 256x128 f32 ref2 column block is staged into
// LDS by the Tensor Data Mover (async DMA, TENSORcnt) before any write: each column
// range is read and written by exactly one WG -> no aliasing hazard.
__global__ __launch_bounds__(256) void ref3_kernel(const float* __restrict__ img2,
                                                   float* __restrict__ outb) {
    __shared__ float Scol[256 * 128];                 // 128 KiB, LDS offset 0
    int blk = blockIdx.x;                             // 896
    int nhalf = blk & 1; int slice = blk >> 1;
    float* O = outb + (size_t)slice * 65536;
    const float* A = img2 + (size_t)slice * 65536;

    // ---- TDM: 2D tile load, tile_dim0=128 elems (x), tile_dim1=256 rows (y),
    //      tensor_dim0_stride=256 elems, data_size=4B, dest LDS addr 0 ----
    {
        unsigned long long ga = (unsigned long long)(uintptr_t)(O + (size_t)nhalf * 128);
        v4u g0;
        g0[0] = 1u;                                   // count=1, user descriptor
        g0[1] = 0u;                                   // lds_addr = 0 (Scol base)
        g0[2] = (unsigned)(ga & 0xFFFFFFFFu);         // global_addr[31:0]
        g0[3] = (unsigned)((ga >> 32) & 0x01FFFFFFu)  // global_addr[56:32]
                | (2u << 30);                         // type = 2 ("image")
        v8u g1;
        g1[0] = (2u << 16);                           // workgroup_mask=0, data_size=4B
        g1[1] = 0u;                                   // tensor_dim0[15:0] = 0 (of 0x10000000)
        g1[2] = 0x1000u;                              // tensor_dim0[31:16], tensor_dim1[15:0]=0
        g1[3] = 0x1000u | (128u << 16);               // tensor_dim1[31:16], tile_dim0=128
        g1[4] = 256u;                                 // tile_dim1=256, tile_dim2=0
        g1[5] = 256u;                                 // tensor_dim0_stride[31:0] = 256
        g1[6] = 0u;                                   // stride0[47:32]=0, stride1[15:0]=0
        g1[7] = 0u;                                   // stride1[47:16]=0
        if (threadIdx.x < 32) {                       // one wave issues the DMA
            asm volatile("tensor_load_to_lds %0, %1" :: "s"(g0), "s"(g1) : "memory");
            asm volatile("s_wait_tensorcnt 0x0" ::: "memory");
        }
    }
    __syncthreads();

    int tid = threadIdx.x;
    int lane = tid & 31, wave = tid >> 5;
    int half = lane >> 4, nlo = lane & 15;
    for (int t = wave; t < 128; t += 8) {             // 16 Mtiles x 8 Ntiles (in half)
        int mt = t >> 3, ntl = t & 7;
        int m = mt * 16 + nlo;
        int n = ntl * 16 + nlo;
        v8f acc = {};
        for (int kt = 0; kt < 8; ++kt) {
            v16h a = loadA_f32(A, m, half, 256, kt * 32);
            v16h bf;
#pragma unroll
            for (int v = 0; v < 8; ++v) {
                int k = kt * 32 + kB(v, half);
                bf[2 * v]     = (_Float16)Scol[k * 128 + n];
                bf[2 * v + 1] = (_Float16)Scol[(k + 1) * 128 + n];
            }
            acc = wmma16(a, bf, acc);
        }
#pragma unroll
        for (int v = 0; v < 8; ++v) {
            int row = mt * 16 + v + 8 * half;
            O[(size_t)row * 256 + nhalf * 128 + n] = Scol[row * 128 + n] + acc[v];
        }
    }
}

// ---------------- host launcher ----------------
extern "C" void kernel_launch(void* const* d_in, const int* in_sizes, int n_in,
                              void* d_out, int out_size, void* d_ws, size_t ws_size,
                              hipStream_t stream) {
    (void)in_sizes; (void)n_in; (void)out_size; (void)ws_size;
    const float* input   = (const float*)d_in[0];
    const float* abun    = (const float*)d_in[1];
    const float* spe_lib = (const float*)d_in[2];
    const float* conv1_w = (const float*)d_in[3];
    const float* conv1_b = (const float*)d_in[4];
    const float* conv3_w = (const float*)d_in[5];
    const float* conv3_b = (const float*)d_in[6];
    float* out = (float*)d_out;

    char* ws = (char*)d_ws;
    float* xfeat = (float*)(ws);
    float* img   = (float*)(ws + WS_XFEAT_BYTES);
    h16*   Bspe  = (h16*)(ws + WS_XFEAT_BYTES + WS_IMG_BYTES);
    h16*   Babun = (h16*)(ws + WS_XFEAT_BYTES + WS_IMG_BYTES + WS_BSPE_HALVES * 2);

    // 1. feature extraction + fragment pre-swizzles (independent)
    conv1_kernel<<<(BSZ * NF * HLR * WLR) / 256, 256, 0, stream>>>(input, conv1_w, conv1_b, xfeat);
    spe_swizzle_kernel<<<((int)WS_BSPE_HALVES + 255) / 256, 256, 0, stream>>>(spe_lib, Bspe);
    abun_swizzle_kernel<<<(int)WS_BABUN_HALVES / 256, 256, 0, stream>>>(abun, Babun);

    // 2. img1 = mlp_pass(offset=0)
    mlp_pass_kernel<<<BSZ * HLR * WLR, 128, 0, stream>>>(xfeat, conv3_w, conv3_b, Bspe, img, 0);
    // 3. d_out = ref2 = img1 @ abun
    ref2_kernel<<<(448 * 16 * 4) / 8, 256, 0, stream>>>(img, Babun, out);
    // 4. img2 = mlp_pass(offset=HALF), reusing img buffer
    mlp_pass_kernel<<<BSZ * HLR * WLR, 128, 0, stream>>>(xfeat, conv3_w, conv3_b, Bspe, img, HALF_OFF);
    // 5. d_out = ref2 + img2 @ ref2
    ref3_kernel<<<448 * 2, 256, 0, stream>>>(img, out);
}
